// CustomGNN_55628416418294
// MI455X (gfx1250) — compile-verified, matrix operand order checked by reading, not run on hardware
//
#include <hip/hip_runtime.h>

#define HIDC 64
#define INC  128

typedef __attribute__((ext_vector_type(16))) __bf16 v16bf;
typedef __attribute__((ext_vector_type(8)))  float  v8f;

// ---------------------------------------------------------------------------
// Fragment loader: builds the CDNA5 16x32 bf16 A/B fragment from row-major f32.
// Lane l (l&15 = row within tile, l>>4 = "half") holds:
//   elements 0..7  : K = kBase + half*8 + j
//   elements 8..15 : K = kBase + 16 + half*8 + j
// (matches CDNA5 ISA 16-bit A-matrix 16x32 layout; B uses the
//  symmetric layout, so loading row n of row-major W gives B[k,n] = W[n,k].)
// ---------------------------------------------------------------------------
__device__ __forceinline__ v16bf load_frag(const float* __restrict__ base, int ld,
                                           int row, int kBase, int lane, float scale) {
  const float* p = base + (size_t)row * ld + kBase + ((lane >> 4) << 3);
  v16bf f;
#pragma unroll
  for (int j = 0; j < 8; ++j) f[j]     = (__bf16)(p[j] * scale);
#pragma unroll
  for (int j = 0; j < 8; ++j) f[8 + j] = (__bf16)(p[16 + j] * scale);
  return f;
}

// D/C layout: lane l -> N = l&15 ; VGPR v -> M = v + 8*(l>>4)
__device__ __forceinline__ void store_tile(float* __restrict__ out, v8f c,
                                           const float* __restrict__ bias,
                                           int tileM, int tileN, int lane, int doRelu) {
  int n = tileN + (lane & 15);
  float b = bias[n];
  int mBase = tileM + ((lane >> 4) << 3);
#pragma unroll
  for (int v = 0; v < 8; ++v) {
    float r = c[v] + b;
    if (doRelu) r = r > 0.f ? r : 0.f;
    out[(size_t)(mBase + v) * HIDC + n] = r;
  }
}

// ---------------------------------------------------------------------------
// fc: out[M,64] = relu(x[M,128] @ W[64,128]^T + b). One wave per 16x16 tile.
// ---------------------------------------------------------------------------
__global__ void fc_relu_wmma(const float* __restrict__ x, const float* __restrict__ W,
                             const float* __restrict__ b, float* __restrict__ out, int M) {
  int wave = blockIdx.x * (blockDim.x >> 5) + (threadIdx.x >> 5);
  int lane = threadIdx.x & 31;
  const int nTilesN = HIDC / 16;                       // 4
  int tileM = (wave / nTilesN) * 16;
  int tileN = (wave % nTilesN) * 16;
  if (tileM >= M) return;                              // wave-uniform
  int rowA = tileM + (lane & 15);
  int rowB = tileN + (lane & 15);
  v8f c = {0.f, 0.f, 0.f, 0.f, 0.f, 0.f, 0.f, 0.f};
#pragma unroll
  for (int kk = 0; kk < INC; kk += 32) {
    v16bf a  = load_frag(x, INC, rowA, kk, lane, 1.0f);
    v16bf bb = load_frag(W, INC, rowB, kk, lane, 1.0f);
    c = __builtin_amdgcn_wmma_f32_16x16x32_bf16(false, a, false, bb, (short)0, c, false, false);
  }
  store_tile(out, c, b, tileM, tileN, lane, 1);
}

// ---------------------------------------------------------------------------
// SAGE combine: out = act( (agg*inv) @ Wl^T + bl + h @ Wr^T ), fused two GEMMs.
// ---------------------------------------------------------------------------
__global__ void sage_combine_wmma(const float* __restrict__ agg, const float* __restrict__ inv,
                                  const float* __restrict__ h,
                                  const float* __restrict__ Wl, const float* __restrict__ bl,
                                  const float* __restrict__ Wr,
                                  float* __restrict__ out, int M, int doRelu) {
  int wave = blockIdx.x * (blockDim.x >> 5) + (threadIdx.x >> 5);
  int lane = threadIdx.x & 31;
  const int nTilesN = HIDC / 16;                       // 4
  int tileM = (wave / nTilesN) * 16;
  int tileN = (wave % nTilesN) * 16;
  if (tileM >= M) return;                              // wave-uniform
  int rowA = tileM + (lane & 15);
  int rowB = tileN + (lane & 15);
  float sA = inv[rowA];                                // 1/max(deg,1) folded into A load
  v8f c = {0.f, 0.f, 0.f, 0.f, 0.f, 0.f, 0.f, 0.f};
#pragma unroll
  for (int kk = 0; kk < HIDC; kk += 32) {
    v16bf a1 = load_frag(agg, HIDC, rowA, kk, lane, sA);
    v16bf b1 = load_frag(Wl,  HIDC, rowB, kk, lane, 1.0f);
    c = __builtin_amdgcn_wmma_f32_16x16x32_bf16(false, a1, false, b1, (short)0, c, false, false);
    v16bf a2 = load_frag(h,   HIDC, rowA, kk, lane, 1.0f);
    v16bf b2 = load_frag(Wr,  HIDC, rowB, kk, lane, 1.0f);
    c = __builtin_amdgcn_wmma_f32_16x16x32_bf16(false, a2, false, b2, (short)0, c, false, false);
  }
  store_tile(out, c, bl, tileM, tileN, lane, doRelu);
}

// ---------------------------------------------------------------------------
// Edge kernels: L2-resident f32 hardware atomics.
// ---------------------------------------------------------------------------
__global__ void zero_f32(float* __restrict__ p, long long n) {
  long long i = (long long)blockIdx.x * blockDim.x + threadIdx.x;
  if (i < n) p[i] = 0.f;
}

__global__ void deg_kernel(const int* __restrict__ dst, float* __restrict__ deg, int E) {
  int e = blockIdx.x * blockDim.x + threadIdx.x;
  if (e < E) unsafeAtomicAdd(deg + dst[e], 1.0f);
}

__global__ void inv_kernel(float* __restrict__ deg, int n) {
  int i = blockIdx.x * blockDim.x + threadIdx.x;
  if (i < n) {
    float d = deg[i];
    deg[i] = 1.0f / (d > 1.0f ? d : 1.0f);
  }
}

// One thread per (edge, channel): lanes 0..63 cover one edge's 256B row.
__global__ void scatter_add(const float* __restrict__ h, const int* __restrict__ src,
                            const int* __restrict__ dst, float* __restrict__ agg, long long E) {
  long long g = (long long)blockIdx.x * blockDim.x + threadIdx.x;
  if (g >= (E << 6)) return;
  int e = (int)(g >> 6);
  int c = (int)(g & 63);
  int s = src[e];
  int d = dst[e];
  unsafeAtomicAdd(agg + (size_t)d * HIDC + c, h[(size_t)s * HIDC + c]);
}

// head: out[m] = dot(h[m,:], w) + b
__global__ void head_kernel(const float* __restrict__ h, const float* __restrict__ w,
                            const float* __restrict__ b, float* __restrict__ out, int M) {
  int m = blockIdx.x * blockDim.x + threadIdx.x;
  if (m >= M) return;
  const float* row = h + (size_t)m * HIDC;
  float acc = b[0];
#pragma unroll
  for (int c = 0; c < HIDC; ++c) acc += row[c] * w[c];
  out[m] = acc;
}

// ---------------------------------------------------------------------------
extern "C" void kernel_launch(void* const* d_in, const int* in_sizes, int n_in,
                              void* d_out, int out_size, void* d_ws, size_t ws_size,
                              hipStream_t stream) {
  const float* x   = (const float*)d_in[0];
  const int*   ei  = (const int*)d_in[1];     // edge_index [2, E]
  const float* fcW = (const float*)d_in[3];
  const float* fcb = (const float*)d_in[4];
  const float* Wl1 = (const float*)d_in[5];
  const float* bl1 = (const float*)d_in[6];
  const float* Wr1 = (const float*)d_in[7];
  const float* Wl2 = (const float*)d_in[8];
  const float* bl2 = (const float*)d_in[9];
  const float* Wr2 = (const float*)d_in[10];
  const float* hW  = (const float*)d_in[11];
  const float* hb  = (const float*)d_in[12];
  float* out = (float*)d_out;

  const int M = in_sizes[0] / INC;            // 100000 (multiple of 16)
  const long long E = in_sizes[1] / 2;        // 3,200,000
  const int* src = ei;
  const int* dst = ei + E;

  float* buf0 = (float*)d_ws;                 // h1, then agg2
  float* buf1 = buf0 + (size_t)M * HIDC;      // agg1, then h3
  float* buf2 = buf1 + (size_t)M * HIDC;      // h2
  float* deg  = buf2 + (size_t)M * HIDC;      // deg -> inv_deg

  const int T = 256;
  int gemmWaves  = (M / 16) * (HIDC / 16);
  int gemmBlocks = (gemmWaves + (T / 32) - 1) / (T / 32);
  long long featN = (long long)M * HIDC;
  int featBlocks  = (int)((featN + T - 1) / T);
  long long scatN = E << 6;
  int scatBlocks  = (int)((scatN + T - 1) / T);

  // degrees (shared by both layers)
  zero_f32<<<(M + T - 1) / T, T, 0, stream>>>(deg, M);
  deg_kernel<<<(int)((E + T - 1) / T), T, 0, stream>>>(dst, deg, (int)E);
  inv_kernel<<<(M + T - 1) / T, T, 0, stream>>>(deg, M);

  // fc + relu
  fc_relu_wmma<<<gemmBlocks, T, 0, stream>>>(x, fcW, fcb, buf0, M);

  // SAGE layer 1
  zero_f32<<<featBlocks, T, 0, stream>>>(buf1, featN);
  scatter_add<<<scatBlocks, T, 0, stream>>>(buf0, src, dst, buf1, E);
  sage_combine_wmma<<<gemmBlocks, T, 0, stream>>>(buf1, deg, buf0, Wl1, bl1, Wr1, buf2, M, 1);

  // SAGE layer 2
  zero_f32<<<featBlocks, T, 0, stream>>>(buf0, featN);
  scatter_add<<<scatBlocks, T, 0, stream>>>(buf2, src, dst, buf0, E);
  sage_combine_wmma<<<gemmBlocks, T, 0, stream>>>(buf0, deg, buf2, Wl2, bl2, Wr2, buf1, M, 0);

  // head
  head_kernel<<<(M + T - 1) / T, T, 0, stream>>>(buf1, hW, hb, out, M);
}
